// GCNModel_76768245449213
// MI455X (gfx1250) — compile-verified, matrix operand order checked by reading, not run on hardware
//
#include <hip/hip_runtime.h>

#define NN 50000
#define EE 800000
#define FIN 256
#define HC 128            // H*C
#define CC 64

typedef __attribute__((ext_vector_type(16))) __bf16 v16bf;
typedef __attribute__((ext_vector_type(8)))  float  v8f;

struct U8 { uint4 a, b; };   // 32 bytes == v16bf

__device__ __forceinline__ unsigned short f2bf(float f) {
    unsigned u = __float_as_uint(f);
    return (unsigned short)((u + 0x7FFFu + ((u >> 16) & 1u)) >> 16);
}
__device__ __forceinline__ unsigned fenc(float f) {
    unsigned u = __float_as_uint(f);
    return (u & 0x80000000u) ? ~u : (u | 0x80000000u);
}
__device__ __forceinline__ float fdec(unsigned x) {
    unsigned u = (x & 0x80000000u) ? (x ^ 0x80000000u) : ~x;
    return __uint_as_float(u);
}
__device__ __forceinline__ float lrelu(float v, float s) { return v > 0.f ? v : s * v; }

// ---------------- prep kernels ----------------
__global__ void prep_wt(const float* __restrict__ W, unsigned short* __restrict__ Wt) {
    int t = blockIdx.x * blockDim.x + threadIdx.x;          // [HC*FIN)
    if (t >= HC * FIN) return;
    int c = t / FIN, k = t % FIN;                           // Wt[c][k] = W[k][c]
    Wt[t] = f2bf(W[(size_t)k * HC + c]);
}

__global__ void prep_bcat(const float* __restrict__ fc1w, unsigned short* __restrict__ Bt) {
    int t = blockIdx.x * blockDim.x + threadIdx.x;          // [HC*HC)
    if (t >= HC * HC) return;
    int c = t / HC, k = t % HC;
    float v = (c < CC) ? fc1w[(size_t)k * CC + c]           // top half  (src part)
                       : fc1w[(size_t)(HC + k) * CC + (c - CC)]; // bottom half (dst part)
    Bt[t] = f2bf(v);
}

__global__ void bn_prep(const float* __restrict__ g, const float* __restrict__ b,
                        const float* __restrict__ mu, const float* __restrict__ var,
                        const float* __restrict__ fc1b,
                        float* __restrict__ S, float* __restrict__ T) {
    int c = threadIdx.x;
    if (c >= CC) return;
    float s = rsqrtf(var[c] + 1e-5f) * g[c];
    S[c] = s;
    T[c] = (fc1b[c] - mu[c]) * s + b[c];
}

// ---------------- bf16 WMMA GEMM:  C[M,128] = A[M,K] * B[K,128] ----------------
// Bt is B transposed+bf16: Bt[col][k], row-major [128][K].
template <int K>
__global__ __launch_bounds__(256) void gemm_bf16(const float* __restrict__ A,
                                                 const unsigned short* __restrict__ Bt,
                                                 float* __restrict__ C, int M) {
    __shared__ __align__(16) unsigned short As[16][K + 8];  // +16B pad: conflict-free b128
    const int row_base = blockIdx.x * 16;
    const int tid = threadIdx.x;

    for (int idx = tid; idx < 16 * K; idx += 256) {
        int r = idx / K, k = idx % K;
        int gr = row_base + r;
        float v = (gr < M) ? A[(size_t)gr * K + k] : 0.f;
        As[r][k] = f2bf(v);
    }
    __syncthreads();

    const int wave = tid >> 5;        // 8 waves -> 8 column tiles of 16
    const int lane = tid & 31;
    const int half = lane >> 4;       // K-range select per ISA A-layout
    const int r    = lane & 15;       // row (A) / col (B) within tile
    const int col_base = wave * 16;

    v8f acc = {};
#pragma unroll
    for (int kk = 0; kk < K / 32; ++kk) {
        U8 ua, ub;
        ua.a = *(const uint4*)&As[r][kk * 32 + half * 8];
        ua.b = *(const uint4*)&As[r][kk * 32 + 16 + half * 8];
        const unsigned short* bp = &Bt[(size_t)(col_base + r) * K + kk * 32 + half * 8];
        ub.a = *(const uint4*)bp;
        ub.b = *(const uint4*)(bp + 16);
        v16bf af = __builtin_bit_cast(v16bf, ua);
        v16bf bf = __builtin_bit_cast(v16bf, ub);
        acc = __builtin_amdgcn_wmma_f32_16x16x32_bf16(false, af, false, bf,
                                                      (short)0, acc, false, false);
    }
#pragma unroll
    for (int i = 0; i < 8; ++i) {
        int row = row_base + i + half * 8;
        if (row < M) C[(size_t)row * HC + col_base + r] = acc[i];
    }
}

// ---------------- attention ----------------
__global__ void att_scores(const float* __restrict__ h, const float* __restrict__ asrc,
                           const float* __restrict__ adst,
                           float* __restrict__ a_s, float* __restrict__ a_d) {
    int t = blockIdx.x * blockDim.x + threadIdx.x;          // [2N)
    if (t >= 2 * NN) return;
    int n = t >> 1, hd = t & 1;
    const float4* hp = (const float4*)(h + (size_t)n * HC + hd * CC);
    const float4* s4 = (const float4*)(asrc + hd * CC);
    const float4* d4 = (const float4*)(adst + hd * CC);
    float s = 0.f, d = 0.f;
#pragma unroll
    for (int q = 0; q < 16; ++q) {
        float4 hv = hp[q], sv = s4[q], dv = d4[q];
        s += hv.x * sv.x + hv.y * sv.y + hv.z * sv.z + hv.w * sv.w;
        d += hv.x * dv.x + hv.y * dv.y + hv.z * dv.z + hv.w * dv.w;
    }
    a_s[t] = s; a_d[t] = d;
}

__device__ __forceinline__ void edge_sd(const int* ei, int e, int& src, int& dst) {
    if (e < EE) { src = ei[e]; dst = ei[EE + e]; }
    else        { src = e - EE; dst = e - EE; }             // self loops
}

__global__ void att_max(const int* __restrict__ ei, const float* __restrict__ a_s,
                        const float* __restrict__ a_d, unsigned* __restrict__ m_ord) {
    int t = blockIdx.x * blockDim.x + threadIdx.x;          // [(E+N)*2)
    if (t >= (EE + NN) * 2) return;
    int e = t >> 1, hd = t & 1, src, dst;
    edge_sd(ei, e, src, dst);
    float v = lrelu(a_s[src * 2 + hd] + a_d[dst * 2 + hd], 0.2f);
    atomicMax(&m_ord[dst * 2 + hd], fenc(v));
}

__global__ void att_sum(const int* __restrict__ ei, const float* __restrict__ a_s,
                        const float* __restrict__ a_d, const unsigned* __restrict__ m_ord,
                        float* __restrict__ denom) {
    int t = blockIdx.x * blockDim.x + threadIdx.x;
    if (t >= (EE + NN) * 2) return;
    int e = t >> 1, hd = t & 1, src, dst;
    edge_sd(ei, e, src, dst);
    float v = lrelu(a_s[src * 2 + hd] + a_d[dst * 2 + hd], 0.2f);
    float ex = expf(v - fdec(m_ord[dst * 2 + hd]));
    __hip_atomic_fetch_add(&denom[dst * 2 + hd], ex,
                           __ATOMIC_RELAXED, __HIP_MEMORY_SCOPE_AGENT);
}

// one wave32 per edge: 32 lanes x float4 = 128 channels; head = lane>>4
__global__ __launch_bounds__(256) void aggregate(const int* __restrict__ ei,
                                                 const float* __restrict__ h,
                                                 const float* __restrict__ a_s,
                                                 const float* __restrict__ a_d,
                                                 const unsigned* __restrict__ m_ord,
                                                 const float* __restrict__ denom,
                                                 float* __restrict__ outp) {
    int e = (blockIdx.x * blockDim.x + threadIdx.x) >> 5;
    if (e >= EE + NN) return;
    int lane = threadIdx.x & 31, hd = lane >> 4, src, dst;
    edge_sd(ei, e, src, dst);
    float v = lrelu(a_s[src * 2 + hd] + a_d[dst * 2 + hd], 0.2f);
    float alpha = expf(v - fdec(m_ord[dst * 2 + hd])) / denom[dst * 2 + hd];
    float4 hv = *(const float4*)(h + (size_t)src * HC + lane * 4);
    float* o = outp + (size_t)dst * HC + lane * 4;
    __hip_atomic_fetch_add(o + 0, hv.x * alpha, __ATOMIC_RELAXED, __HIP_MEMORY_SCOPE_AGENT);
    __hip_atomic_fetch_add(o + 1, hv.y * alpha, __ATOMIC_RELAXED, __HIP_MEMORY_SCOPE_AGENT);
    __hip_atomic_fetch_add(o + 2, hv.z * alpha, __ATOMIC_RELAXED, __HIP_MEMORY_SCOPE_AGENT);
    __hip_atomic_fetch_add(o + 3, hv.w * alpha, __ATOMIC_RELAXED, __HIP_MEMORY_SCOPE_AGENT);
}

__global__ void postact(float* __restrict__ outp, const float* __restrict__ cb) {
    int t = blockIdx.x * blockDim.x + threadIdx.x;
    if (t >= NN * HC) return;
    outp[t] = lrelu(outp[t] + cb[t & (HC - 1)], 0.01f);
}

// per-edge: z = u[src]+v[dst]; z=z*S+T; leaky; out = z @ fc2 + b2
__global__ __launch_bounds__(256) void edge_final(const int* __restrict__ ei,
                                                  const float* __restrict__ uv,
                                                  const float* __restrict__ S,
                                                  const float* __restrict__ T,
                                                  const float* __restrict__ w2,
                                                  const float* __restrict__ b2,
                                                  float* __restrict__ out) {
    __shared__ float Sl[CC], Tl[CC], W2l[CC * 3], B2l[3];
    int tid = threadIdx.x;
    if (tid < CC) { Sl[tid] = S[tid]; Tl[tid] = T[tid]; }
    if (tid < CC * 3) W2l[tid] = w2[tid];
    if (tid < 3) B2l[tid] = b2[tid];
    __syncthreads();

    int e = blockIdx.x * blockDim.x + tid;
    if (e >= EE) return;
    int src = ei[e], dst = ei[EE + e];
    const float4* U4 = (const float4*)(uv + (size_t)src * HC);        // u part
    const float4* V4 = (const float4*)(uv + (size_t)dst * HC + CC);   // v part
    float o0 = 0.f, o1 = 0.f, o2 = 0.f;
#pragma unroll
    for (int q = 0; q < 16; ++q) {
        float4 a = U4[q], b = V4[q];
        float zz[4] = {a.x + b.x, a.y + b.y, a.z + b.z, a.w + b.w};
#pragma unroll
        for (int j = 0; j < 4; ++j) {
            int c = q * 4 + j;
            float z = lrelu(zz[j] * Sl[c] + Tl[c], 0.01f);
            o0 += z * W2l[c * 3 + 0];
            o1 += z * W2l[c * 3 + 1];
            o2 += z * W2l[c * 3 + 2];
        }
    }
    out[(size_t)e * 3 + 0] = o0 + B2l[0];
    out[(size_t)e * 3 + 1] = o1 + B2l[1];
    out[(size_t)e * 3 + 2] = o2 + B2l[2];
}

// ---------------- host ----------------
static char* wsal(char*& p, size_t bytes) {
    char* r = p;
    p += (bytes + 255) & ~(size_t)255;
    return r;
}

extern "C" void kernel_launch(void* const* d_in, const int* in_sizes, int n_in,
                              void* d_out, int out_size, void* d_ws, size_t ws_size,
                              hipStream_t stream) {
    (void)in_sizes; (void)n_in; (void)out_size; (void)ws_size;
    const float* x      = (const float*)d_in[0];
    const int*   ei     = (const int*)d_in[1];
    const float* W      = (const float*)d_in[2];
    const float* asrc   = (const float*)d_in[3];
    const float* adst   = (const float*)d_in[4];
    const float* cbias  = (const float*)d_in[5];
    const float* fc1w   = (const float*)d_in[6];
    const float* fc1b   = (const float*)d_in[7];
    const float* bng    = (const float*)d_in[8];
    const float* bnb    = (const float*)d_in[9];
    const float* bnm    = (const float*)d_in[10];
    const float* bnv    = (const float*)d_in[11];
    const float* fc2w   = (const float*)d_in[12];
    const float* fc2b   = (const float*)d_in[13];
    float* out = (float*)d_out;

    char* p = (char*)d_ws;
    float*          h     = (float*)wsal(p, (size_t)NN * HC * 4);
    float*          agg   = (float*)wsal(p, (size_t)NN * HC * 4);
    float*          uv    = (float*)wsal(p, (size_t)NN * HC * 4);
    float*          a_s   = (float*)wsal(p, (size_t)NN * 2 * 4);
    float*          a_d   = (float*)wsal(p, (size_t)NN * 2 * 4);
    unsigned*       m_ord = (unsigned*)wsal(p, (size_t)NN * 2 * 4);
    float*          denom = (float*)wsal(p, (size_t)NN * 2 * 4);
    unsigned short* Wt    = (unsigned short*)wsal(p, (size_t)HC * FIN * 2);
    unsigned short* Bct   = (unsigned short*)wsal(p, (size_t)HC * HC * 2);
    float*          Sb    = (float*)wsal(p, CC * 4);
    float*          Tb    = (float*)wsal(p, CC * 4);

    const int EA = EE + NN;

    prep_wt<<<(HC * FIN + 255) / 256, 256, 0, stream>>>(W, Wt);
    prep_bcat<<<(HC * HC + 255) / 256, 256, 0, stream>>>(fc1w, Bct);
    bn_prep<<<1, 64, 0, stream>>>(bng, bnb, bnm, bnv, fc1b, Sb, Tb);

    hipMemsetAsync(m_ord, 0, (size_t)NN * 2 * 4, stream);
    hipMemsetAsync(denom, 0, (size_t)NN * 2 * 4, stream);
    hipMemsetAsync(agg, 0, (size_t)NN * HC * 4, stream);

    gemm_bf16<FIN><<<(NN + 15) / 16, 256, 0, stream>>>(x, Wt, h, NN);
    att_scores<<<(2 * NN + 255) / 256, 256, 0, stream>>>(h, asrc, adst, a_s, a_d);
    att_max<<<(EA * 2 + 255) / 256, 256, 0, stream>>>(ei, a_s, a_d, m_ord);
    att_sum<<<(EA * 2 + 255) / 256, 256, 0, stream>>>(ei, a_s, a_d, m_ord, denom);
    aggregate<<<((size_t)EA * 32 + 255) / 256, 256, 0, stream>>>(ei, h, a_s, a_d,
                                                                 m_ord, denom, agg);
    postact<<<(NN * HC + 255) / 256, 256, 0, stream>>>(agg, cbias);
    gemm_bf16<HC><<<(NN + 15) / 16, 256, 0, stream>>>(agg, Bct, uv, NN);
    edge_final<<<(EE + 255) / 256, 256, 0, stream>>>(ei, uv, Sb, Tb, fc2w, fc2b, out);
}
